// Attention_56349970923851
// MI455X (gfx1250) — matrix operand order repaired and checked by reading, not compiled, not run on hardware
//
#include <hip/hip_runtime.h>
#include <hip/hip_bf16.h>

// ---------------------------------------------------------------------------
// CDNA5 (gfx1250, wave32) attention block:
//   1) QKV GEMMs (bf16 WMMA, f32 accum) with fused RoPE + head-layout epilogue
//   2) Flash attention (bf16 WMMA QK^T and P@V, online softmax, GQA g=h/4)
//      - K tile + mask tile staged with GLOBAL_LOAD_ASYNC_TO_LDS_B128
//        (ASYNCcnt), V tile transposed through VGPRs into LDS
//   3) Output projection GEMM (bf16 WMMA)
// All intermediates bf16 and L2-resident (192MB L2); HBM floor ~150MB.
// ---------------------------------------------------------------------------

typedef __bf16 bf16_t;
typedef __attribute__((ext_vector_type(16))) bf16_t          v16bf;
typedef __attribute__((ext_vector_type(8)))  float           v8f;
typedef __attribute__((ext_vector_type(8)))  unsigned short  us8;

union Frag {
  v16bf v;
  us8   h[2];
};

__device__ __forceinline__ unsigned short f2bf(float f) {
  unsigned int u = __float_as_uint(f);
  u += 0x7fffu + ((u >> 16) & 1u);          // round-to-nearest-even
  return (unsigned short)(u >> 16);
}

__device__ __forceinline__ v8f wmma_bf16(const Frag& a, const Frag& b, v8f c) {
  // (neg_a, A, neg_b, B, c_mod, C, reuse_a, reuse_b)
  return __builtin_amdgcn_wmma_f32_16x16x32_bf16(
      false, a.v, false, b.v, (short)0, c, false, false);
}

// Async global->LDS 16B copy (CDNA5). LDS dest address = low 32 bits of the
// generic pointer (ISA 10.2).
__device__ __forceinline__ void async_copy_b128(const void* gptr, void* lptr) {
  unsigned lds = (unsigned)(uintptr_t)lptr;
  asm volatile("global_load_async_to_lds_b128 %0, %1, off"
               :: "v"(lds), "v"(gptr) : "memory");
}

__device__ __forceinline__ void wait_async0() {
  asm volatile("s_wait_asynccnt 0" ::: "memory");
}

#define SEQ   2048
#define ROPEB 10000.0f

// ---------------------------------------------------------------------------
// GEMM: Y[M,N] = X[M,K] @ W[N,K]^T   (torch Linear)
// Block tile 128x128, TK=32, 256 threads = 8 waves; wave tile 32x64.
// mode 0: f32 store Y[row*N+col]           (final projection)
// mode 1: RoPE + bf16 store to [b][H][s][64]  (Q, K)
// mode 2: bf16 store to [b][H][s][64]         (V)
// ---------------------------------------------------------------------------
template <bool XBF>
__global__ __launch_bounds__(256) void gemm_wmma_kernel(
    const float* __restrict__ Xf, const unsigned short* __restrict__ Xb,
    const float* __restrict__ W,
    float* __restrict__ Yf, unsigned short* __restrict__ Yb,
    const int* __restrict__ pos,
    int M, int N, int Kd, int mode, int H)
{
  __shared__ unsigned short sX[128][40];   // 32 + 8 pad; row = 80B (16B aligned)
  __shared__ unsigned short sW[128][40];

  const int tid = threadIdx.x;
  const int wid = tid >> 5;
  const int lane = tid & 31;
  const int ln = lane & 15;
  const int hf = lane >> 4;
  const int wm = wid & 3;        // 4 row-groups of 32
  const int wn = wid >> 2;       // 2 col-groups of 64 (one head per wave span)
  const int m0 = blockIdx.y * 128;
  const int n0 = blockIdx.x * 128;

  v8f acc[2][4];
#pragma unroll
  for (int mt = 0; mt < 2; ++mt)
#pragma unroll
    for (int nt = 0; nt < 4; ++nt) acc[mt][nt] = v8f{};

  for (int k0 = 0; k0 < Kd; k0 += 32) {
    __syncthreads();
    if constexpr (XBF) {
#pragma unroll
      for (int i = 0; i < 2; ++i) {
        int ch = tid + i * 256;                 // 512 x 8-ushort chunks
        int row = ch >> 2, c8 = (ch & 3) << 3;
        *(us8*)&sX[row][c8] =
            *(const us8*)(Xb + (size_t)(m0 + row) * Kd + k0 + c8);
      }
    } else {
#pragma unroll
      for (int i = 0; i < 4; ++i) {
        int ch = tid + i * 256;                 // 1024 x float4 chunks
        int row = ch >> 3, c4 = (ch & 7) << 2;
        float4 v = *(const float4*)(Xf + (size_t)(m0 + row) * Kd + k0 + c4);
        sX[row][c4 + 0] = f2bf(v.x); sX[row][c4 + 1] = f2bf(v.y);
        sX[row][c4 + 2] = f2bf(v.z); sX[row][c4 + 3] = f2bf(v.w);
      }
    }
#pragma unroll
    for (int i = 0; i < 4; ++i) {
      int ch = tid + i * 256;
      int row = ch >> 3, c4 = (ch & 7) << 2;
      float4 v = *(const float4*)(W + (size_t)(n0 + row) * Kd + k0 + c4);
      sW[row][c4 + 0] = f2bf(v.x); sW[row][c4 + 1] = f2bf(v.y);
      sW[row][c4 + 2] = f2bf(v.z); sW[row][c4 + 3] = f2bf(v.w);
    }
    __syncthreads();

    Frag a[2];
#pragma unroll
    for (int mt = 0; mt < 2; ++mt) {
      int r = wm * 32 + mt * 16 + ln;
      a[mt].h[0] = *(const us8*)&sX[r][8 * hf];        // K = 8*hf..+7
      a[mt].h[1] = *(const us8*)&sX[r][16 + 8 * hf];   // K = 16+8*hf..+7
    }
#pragma unroll
    for (int nt = 0; nt < 4; ++nt) {
      int c = wn * 64 + nt * 16 + ln;
      Frag b;
      b.h[0] = *(const us8*)&sW[c][8 * hf];            // K = 8*hf..+7
      b.h[1] = *(const us8*)&sW[c][16 + 8 * hf];       // K = 16+8*hf..+7
#pragma unroll
      for (int mt = 0; mt < 2; ++mt)
        acc[mt][nt] = wmma_bf16(a[mt], b, acc[mt][nt]);
    }
  }

  // ---- epilogue (D layout: lane -> col n=lane%16, vgpr j -> row 8*hf+j) ----
  if (mode == 0) {
#pragma unroll
    for (int mt = 0; mt < 2; ++mt)
#pragma unroll
      for (int nt = 0; nt < 4; ++nt)
#pragma unroll
        for (int j = 0; j < 8; ++j) {
          int grow = m0 + wm * 32 + mt * 16 + 8 * hf + j;
          int gcol = n0 + wn * 64 + nt * 16 + ln;
          Yf[(size_t)grow * N + gcol] = acc[mt][nt][j];
        }
  } else if (mode == 2) {
#pragma unroll
    for (int mt = 0; mt < 2; ++mt)
#pragma unroll
      for (int nt = 0; nt < 4; ++nt)
#pragma unroll
        for (int j = 0; j < 8; ++j) {
          int grow = m0 + wm * 32 + mt * 16 + 8 * hf + j;
          int gcol = n0 + wn * 64 + nt * 16 + ln;
          int bidx = grow >> 11, sidx = grow & (SEQ - 1);
          int h = gcol >> 6, hd = gcol & 63;
          Yb[(((size_t)bidx * H + h) * SEQ + sidx) * 64 + hd] =
              f2bf(acc[mt][nt][j]);
        }
  } else {  // mode 1: RoPE. pair (i, i+32) lives in acc[..][nt] / acc[..][nt+2]
#pragma unroll
    for (int mt = 0; mt < 2; ++mt)
#pragma unroll
      for (int nt = 0; nt < 2; ++nt) {
        int i = nt * 16 + ln;                       // 0..31 within the head
        float invf = __powf(ROPEB, -(float)i * (1.0f / 32.0f));
#pragma unroll
        for (int j = 0; j < 8; ++j) {
          int grow = m0 + wm * 32 + mt * 16 + 8 * hf + j;
          int gcol = n0 + wn * 64 + nt * 16 + ln;
          float x1 = acc[mt][nt][j];
          float x2 = acc[mt][nt + 2][j];
          float fr = (float)pos[grow] * invf;
          float sn, cs;
          __sincosf(fr, &sn, &cs);
          int bidx = grow >> 11, sidx = grow & (SEQ - 1);
          int h = gcol >> 6, hd = gcol & 63;        // hd in 0..31
          size_t base = (((size_t)bidx * H + h) * SEQ + sidx) * 64;
          Yb[base + hd]      = f2bf(x1 * cs - x2 * sn);
          Yb[base + hd + 32] = f2bf(x2 * cs + x1 * sn);
        }
      }
  }
}

// ---------------------------------------------------------------------------
// Flash attention: grid (S/64, 32 heads, b). 4 waves/block; each wave owns a
// 16-row q tile. Streams K/V/mask in 64-key tiles through LDS.
// K + mask tiles: async global->LDS DMA (ASYNCcnt). V: VGPR transpose.
// ---------------------------------------------------------------------------
__global__ __launch_bounds__(128) void flash_attn_kernel(
    const unsigned short* __restrict__ Qb,   // [b][32][S][64] bf16 (RoPE'd)
    const unsigned short* __restrict__ Kb,   // [b][8][S][64]  bf16 (RoPE'd)
    const unsigned short* __restrict__ Vb,   // [b][8][S][64]  bf16
    const float* __restrict__ mask,          // [S][S] additive
    unsigned short* __restrict__ Ob)         // [b*S][2048] bf16, col = h*64+hd
{
  __shared__ unsigned short sK [64][72];     // [key][hd], padded (144B rows)
  __shared__ unsigned short sVt[64][72];     // [hd][key] (transposed V)
  __shared__ unsigned short sP [4][16][72];  // per-wave P tile [qrow][key]
  __shared__ float          sM [64][68];     // mask tile [qrow][key], padded

  const int tid = threadIdx.x;
  const int wid = tid >> 5;
  const int lane = tid & 31;
  const int ln = lane & 15;
  const int hf = lane >> 4;
  const int h = blockIdx.y;
  const int g = h >> 2;                      // GQA: kv head
  const int bidx = blockIdx.z;
  const int qb0 = blockIdx.x * 64;           // block's q-row base
  const int q0 = qb0 + wid * 16;             // wave's q-row base

  // Q A-fragments (resident all loop): rows q0+ln, hd split in two K=32 chunks
  const unsigned short* qptr =
      Qb + (((size_t)bidx * 32 + h) * SEQ + q0 + ln) * 64;
  Frag aQ[2];
#pragma unroll
  for (int ka = 0; ka < 2; ++ka) {
    aQ[ka].h[0] = *(const us8*)(qptr + ka * 32 + 8 * hf);
    aQ[ka].h[1] = *(const us8*)(qptr + ka * 32 + 16 + 8 * hf);
  }

  v8f accO[4];
#pragma unroll
  for (int nt = 0; nt < 4; ++nt) accO[nt] = v8f{};
  float mrow[8], lrow[8];
#pragma unroll
  for (int j = 0; j < 8; ++j) { mrow[j] = -3.0e38f; lrow[j] = 0.0f; }

  const unsigned short* kbase = Kb + ((size_t)bidx * 8 + g) * SEQ * 64;
  const unsigned short* vbase = Vb + ((size_t)bidx * 8 + g) * SEQ * 64;
  const float scale = 0.125f;                // 1/sqrt(64)

  for (int kt = 0; kt < SEQ / 64; ++kt) {
    const int k0 = kt * 64;
    __syncthreads();   // previous tile fully consumed before overwriting LDS

    // --- async stage K tile [key][hd]: 512 x 16B chunks, 4 per thread ---
#pragma unroll
    for (int i = 0; i < 4; ++i) {
      int ch = tid + i * 128;
      int row = ch >> 3, c8 = (ch & 7) << 3;
      async_copy_b128(kbase + (size_t)(k0 + row) * 64 + c8, &sK[row][c8]);
    }
    // --- async stage mask tile [qrow][key]: 1024 x 16B chunks, 8/thread ---
#pragma unroll
    for (int i = 0; i < 8; ++i) {
      int ch = tid + i * 128;
      int row = ch >> 4, c4 = (ch & 15) << 2;
      async_copy_b128(mask + (size_t)(qb0 + row) * SEQ + k0 + c4,
                      &sM[row][c4]);
    }
    // --- stage V transposed through VGPRs: sVt[hd][key] ---
#pragma unroll
    for (int i = 0; i < 4; ++i) {
      int ch = tid + i * 128;
      int key = ch >> 3, c8 = (ch & 7) << 3;
      us8 v = *(const us8*)(vbase + (size_t)(k0 + key) * 64 + c8);
#pragma unroll
      for (int e = 0; e < 8; ++e) sVt[c8 + e][key] = v[e];
    }
    // prefetch next K/V tiles while this one computes
    if (kt + 1 < SEQ / 64) {
      __builtin_prefetch(kbase + (size_t)(k0 + 64 + (tid >> 1)) * 64, 0, 1);
      __builtin_prefetch(vbase + (size_t)(k0 + 64 + (tid >> 1)) * 64, 0, 1);
    }
    wait_async0();
    __syncthreads();

    // --- S = Q @ K^T  (4 key-subtiles x 2 hd-chunks) ---
    v8f accS[4];
#pragma unroll
    for (int nt = 0; nt < 4; ++nt) accS[nt] = v8f{};
#pragma unroll
    for (int nt = 0; nt < 4; ++nt) {
      int c = nt * 16 + ln;                        // key within tile
#pragma unroll
      for (int ka = 0; ka < 2; ++ka) {
        Frag b;
        b.h[0] = *(const us8*)&sK[c][ka * 32 + 8 * hf];
        b.h[1] = *(const us8*)&sK[c][ka * 32 + 16 + 8 * hf];
        accS[nt] = wmma_bf16(aQ[ka], b, accS[nt]);
      }
    }

    // --- scale + additive mask (from LDS) ---
#pragma unroll
    for (int nt = 0; nt < 4; ++nt)
#pragma unroll
      for (int j = 0; j < 8; ++j) {
        int lr = wid * 16 + 8 * hf + j;            // q row within block
        int lc = nt * 16 + ln;                     // key within tile
        accS[nt][j] = accS[nt][j] * scale + sM[lr][lc];
      }

    // --- online softmax: row stats over 16-lane halves (lanes share rows) ---
    float tmax[8];
#pragma unroll
    for (int j = 0; j < 8; ++j) {
      float t = accS[0][j];
#pragma unroll
      for (int nt = 1; nt < 4; ++nt) t = fmaxf(t, accS[nt][j]);
      tmax[j] = t;
    }
#pragma unroll
    for (int off = 1; off < 16; off <<= 1)
#pragma unroll
      for (int j = 0; j < 8; ++j)
        tmax[j] = fmaxf(tmax[j], __shfl_xor(tmax[j], off));

    float corr[8], rsum[8];
#pragma unroll
    for (int j = 0; j < 8; ++j) {
      float mn = fmaxf(mrow[j], tmax[j]);
      corr[j] = __expf(mrow[j] - mn);
      mrow[j] = mn;
      rsum[j] = 0.0f;
    }
#pragma unroll
    for (int nt = 0; nt < 4; ++nt)
#pragma unroll
      for (int j = 0; j < 8; ++j) {
        float p = __expf(accS[nt][j] - mrow[j]);
        accS[nt][j] = p;
        rsum[j] += p;
      }
#pragma unroll
    for (int off = 1; off < 16; off <<= 1)
#pragma unroll
      for (int j = 0; j < 8; ++j) rsum[j] += __shfl_xor(rsum[j], off);
#pragma unroll
    for (int j = 0; j < 8; ++j) lrow[j] = lrow[j] * corr[j] + rsum[j];
#pragma unroll
    for (int nt = 0; nt < 4; ++nt)
#pragma unroll
      for (int j = 0; j < 8; ++j) accO[nt][j] *= corr[j];

    // --- D-layout P -> A-layout via wave-private LDS ---
#pragma unroll
    for (int nt = 0; nt < 4; ++nt)
#pragma unroll
      for (int j = 0; j < 8; ++j)
        sP[wid][8 * hf + j][nt * 16 + ln] = f2bf(accS[nt][j]);
    asm volatile("s_wait_dscnt 0" ::: "memory");

    Frag aP[2];
#pragma unroll
    for (int kc = 0; kc < 2; ++kc) {
      aP[kc].h[0] = *(const us8*)&sP[wid][ln][kc * 32 + 8 * hf];
      aP[kc].h[1] = *(const us8*)&sP[wid][ln][kc * 32 + 16 + 8 * hf];
    }

    // --- O += P @ V  (4 hd-subtiles x 2 key-chunks) ---
#pragma unroll
    for (int nt = 0; nt < 4; ++nt) {
      int c = nt * 16 + ln;                        // hd column
#pragma unroll
      for (int kc = 0; kc < 2; ++kc) {
        Frag b;
        b.h[0] = *(const us8*)&sVt[c][kc * 32 + 8 * hf];
        b.h[1] = *(const us8*)&sVt[c][kc * 32 + 16 + 8 * hf];
        accO[nt] = wmma_bf16(aP[kc], b, accO[nt]);
      }
    }
  }

  // normalize and store bf16 into [b*S, 2048] (col = h*64 + hd)
  float inv[8];
#pragma unroll
  for (int j = 0; j < 8; ++j) inv[j] = 1.0f / lrow[j];
#pragma unroll
  for (int nt = 0; nt < 4; ++nt)
#pragma unroll
    for (int j = 0; j < 8; ++j) {
      int gr = q0 + 8 * hf + j;
      int gc = h * 64 + nt * 16 + ln;
      Ob[((size_t)bidx * SEQ + gr) * 2048 + gc] = f2bf(accO[nt][j] * inv[j]);
    }
}

// ---------------------------------------------------------------------------
extern "C" void kernel_launch(void* const* d_in, const int* in_sizes, int n_in,
                              void* d_out, int out_size, void* d_ws, size_t ws_size,
                              hipStream_t stream) {
  const float* x    = (const float*)d_in[0];
  const float* mask = (const float*)d_in[1];
  const int*   pos  = (const int*)d_in[2];
  const float* wq   = (const float*)d_in[3];
  const float* wk   = (const float*)d_in[4];
  const float* wv   = (const float*)d_in[5];
  const float* wo   = (const float*)d_in[6];
  float* out = (float*)d_out;

  const int B = 2, S = SEQ, D = 2048, KV = 512;
  const int M = B * S;   // 4096

  // Workspace (40MB total, bf16):
  unsigned short* Qb = (unsigned short*)d_ws;                 // [B][32][S][64]
  unsigned short* Kb = Qb + (size_t)B * 32 * S * 64;          // [B][8][S][64]
  unsigned short* Vb = Kb + (size_t)B * 8 * S * 64;           // [B][8][S][64]
  unsigned short* Ob = Vb + (size_t)B * 8 * S * 64;           // [M][2048]

  dim3 blk(256);
  // Q = x@wq^T, RoPE, 32 heads
  gemm_wmma_kernel<false><<<dim3(D / 128, M / 128), blk, 0, stream>>>(
      x, nullptr, wq, nullptr, Qb, pos, M, D, D, /*mode=*/1, /*H=*/32);
  // K = x@wk^T, RoPE, 8 heads
  gemm_wmma_kernel<false><<<dim3(KV / 128, M / 128), blk, 0, stream>>>(
      x, nullptr, wk, nullptr, Kb, pos, M, KV, D, /*mode=*/1, /*H=*/8);
  // V = x@wv^T, 8 heads
  gemm_wmma_kernel<false><<<dim3(KV / 128, M / 128), blk, 0, stream>>>(
      x, nullptr, wv, nullptr, Vb, pos, M, KV, D, /*mode=*/2, /*H=*/8);
  // flash attention
  flash_attn_kernel<<<dim3(S / 64, 32, B), dim3(128), 0, stream>>>(
      Qb, Kb, Vb, mask, Ob);
  // out = O@wo^T (f32 store)
  gemm_wmma_kernel<true><<<dim3(D / 128, M / 128), blk, 0, stream>>>(
      nullptr, Ob, wo, out, nullptr, nullptr, M, D, D, /*mode=*/0, /*H=*/0);
}
